// Attention_53798760350071
// MI455X (gfx1250) — matrix operand order repaired and checked by reading, not compiled, not run on hardware
//
#include <hip/hip_runtime.h>

typedef __attribute__((ext_vector_type(16))) _Float16 v16h;
typedef __attribute__((ext_vector_type(8)))  _Float16 v8h;
typedef __attribute__((ext_vector_type(8)))  float    v8f;

#define BQ   4
#define SEQ  2048
#define DIM  768
#define NH   12
#define HD   64
#define BHN  (BQ*NH)        // 48
#define MROWS (BQ*SEQ)      // 8192

#define TSTR 40             // padded f16 stride for 32-wide GEMM tiles (80B)
#define PSTR 72             // padded f16 stride for 64-wide P scratch (144B)

#if defined(__has_builtin)
#if __has_builtin(__builtin_amdgcn_global_load_async_to_lds_b128) && \
    __has_builtin(__builtin_amdgcn_s_wait_asynccnt)
#define USE_ASYNC_LDS 1
#endif
#endif
#ifndef USE_ASYNC_LDS
#define USE_ASYNC_LDS 0
#endif

#if USE_ASYNC_LDS
// exact parameter types per hipcc diagnostic: vector_size(16) int, AS1 / AS3
typedef int vsi4 __attribute__((vector_size(16)));
typedef __attribute__((address_space(1))) vsi4 gvsi4;
typedef __attribute__((address_space(3))) vsi4 lvsi4;
#endif

// ---------------------------------------------------------------------------
// cross-lane helpers: ds_swizzle XOR within 16-lane groups (no LDS memory)
// imm = (xor_mask<<10) | and_mask(0x1f)
// ---------------------------------------------------------------------------
__device__ __forceinline__ float xor_max16(float v) {
    v = fmaxf(v, __int_as_float(__builtin_amdgcn_ds_swizzle(__float_as_int(v), (1 << 10) | 0x1f)));
    v = fmaxf(v, __int_as_float(__builtin_amdgcn_ds_swizzle(__float_as_int(v), (2 << 10) | 0x1f)));
    v = fmaxf(v, __int_as_float(__builtin_amdgcn_ds_swizzle(__float_as_int(v), (4 << 10) | 0x1f)));
    v = fmaxf(v, __int_as_float(__builtin_amdgcn_ds_swizzle(__float_as_int(v), (8 << 10) | 0x1f)));
    return v;
}
__device__ __forceinline__ float xor_sum16(float v) {
    v += __int_as_float(__builtin_amdgcn_ds_swizzle(__float_as_int(v), (1 << 10) | 0x1f));
    v += __int_as_float(__builtin_amdgcn_ds_swizzle(__float_as_int(v), (2 << 10) | 0x1f));
    v += __int_as_float(__builtin_amdgcn_ds_swizzle(__float_as_int(v), (4 << 10) | 0x1f));
    v += __int_as_float(__builtin_amdgcn_ds_swizzle(__float_as_int(v), (8 << 10) | 0x1f));
    return v;
}

// ---------------------------------------------------------------------------
// fp32 -> f16 convert / transpose-convert
// ---------------------------------------------------------------------------
__global__ void cvt_f16(const float* __restrict__ src, _Float16* __restrict__ dst, int n) {
    int i = blockIdx.x * blockDim.x + threadIdx.x;
    if (i < n) dst[i] = (_Float16)src[i];
}
__global__ void cvtT_f16(const float* __restrict__ src, _Float16* __restrict__ dst, int K, int Nn) {
    int i = blockIdx.x * blockDim.x + threadIdx.x;
    if (i < K * Nn) {
        int n = i / K;
        int k = i - n * K;
        dst[i] = (_Float16)src[(size_t)k * Nn + n];
    }
}

// ---------------------------------------------------------------------------
// WMMA 16x16x32 f16 fragment helpers (ISA 7.12.2, wave32)
// A frag: lane<16 -> M=lane, K={0..15}; lane>=16 -> K={16..31}
// B frag: elements 0..15 = K ascending from kbase=16*(lane>=16), col N=lane&15
// ---------------------------------------------------------------------------
__device__ __forceinline__ v16h load_a_frag(const _Float16* rowp, int koff) {
    v8h lo = *(const v8h*)(rowp + 2 * koff);
    v8h hi = *(const v8h*)(rowp + 2 * koff + 8);
    v16h r;
#pragma unroll
    for (int i = 0; i < 8; ++i) { r[i] = lo[i]; r[8 + i] = hi[i]; }
    return r;
}
__device__ __forceinline__ v16h load_b_frag2(const _Float16* p) {  // 16B-aligned pair
    v8h lo = *(const v8h*)(p);
    v8h hi = *(const v8h*)(p + 8);
    v16h r;
#pragma unroll
    for (int i = 0; i < 8; ++i) { r[i] = lo[i]; r[8 + i] = hi[i]; }
    return r;
}

// ---------------------------------------------------------------------------
// 128x128x32-tile GEMM, C = A[f16 MxK] * Bt[f16 NxK]^T, WMMA fp32 accum.
// MODE 0: QKV epilogue (scatter to head-major q/k and transposed V)
// MODE 1: proj epilogue (fp32 out + bias)
// ---------------------------------------------------------------------------
template <int MODE>
__global__ __launch_bounds__(256) void gemm128(
    const _Float16* __restrict__ A, const _Float16* __restrict__ Bt,
    int K, int Nout,
    _Float16* __restrict__ qh, _Float16* __restrict__ kh, _Float16* __restrict__ vTh,
    float* __restrict__ out, const float* __restrict__ bias)
{
    __shared__ _Float16 As[128 * TSTR];
    __shared__ _Float16 Bs[128 * TSTR];

    const int tid  = threadIdx.x;
    const int lane = tid & 31;
    const int w    = tid >> 5;
    const int wm   = w >> 2;          // 0..1
    const int wn   = w & 3;           // 0..3
    const int lm   = lane & 15;
    const int lh   = lane >> 4;
    const int m0   = blockIdx.y * 128;
    const int n0   = blockIdx.x * 128;

    const int ldrow = tid >> 1;            // 0..127
    const int ldseg = (tid & 1) * 16;      // 0 / 16 (f16 units)

    v8f acc[4][2] = {};

    for (int kk = 0; kk < K; kk += 32) {
        const _Float16* gA = A + (size_t)(m0 + ldrow) * K + kk + ldseg;
        const _Float16* gB = Bt + (size_t)(n0 + ldrow) * K + kk + ldseg;
        _Float16* sA = As + ldrow * TSTR + ldseg;
        _Float16* sB = Bs + ldrow * TSTR + ldseg;
#if USE_ASYNC_LDS
        __builtin_amdgcn_global_load_async_to_lds_b128((gvsi4*)gA,       (lvsi4*)sA,       0, 0);
        __builtin_amdgcn_global_load_async_to_lds_b128((gvsi4*)(gA + 8), (lvsi4*)(sA + 8), 0, 0);
        __builtin_amdgcn_global_load_async_to_lds_b128((gvsi4*)gB,       (lvsi4*)sB,       0, 0);
        __builtin_amdgcn_global_load_async_to_lds_b128((gvsi4*)(gB + 8), (lvsi4*)(sB + 8), 0, 0);
        __builtin_amdgcn_s_wait_asynccnt(0);
#else
        *(uint4*)(sA)     = *(const uint4*)(gA);
        *(uint4*)(sA + 8) = *(const uint4*)(gA + 8);
        *(uint4*)(sB)     = *(const uint4*)(gB);
        *(uint4*)(sB + 8) = *(const uint4*)(gB + 8);
        if (kk + 32 < K) {
            __builtin_prefetch(gA + 32, 0, 1);
            __builtin_prefetch(gB + 32, 0, 1);
        }
#endif
        __syncthreads();

        const int koff = lh * 8;
        v16h af[4];
#pragma unroll
        for (int mi = 0; mi < 4; ++mi)
            af[mi] = load_a_frag(As + (wm * 64 + mi * 16 + lm) * TSTR, koff);
        v16h bf[2];
#pragma unroll
        for (int ni = 0; ni < 2; ++ni)
            bf[ni] = load_b_frag2(Bs + (wn * 32 + ni * 16 + lm) * TSTR + lh * 16);

#pragma unroll
        for (int mi = 0; mi < 4; ++mi)
#pragma unroll
            for (int ni = 0; ni < 2; ++ni)
                acc[mi][ni] = __builtin_amdgcn_wmma_f32_16x16x32_f16(
                    false, af[mi], false, bf[ni], (short)0, acc[mi][ni], false, false);

        __syncthreads();
    }

#pragma unroll
    for (int mi = 0; mi < 4; ++mi) {
#pragma unroll
        for (int ni = 0; ni < 2; ++ni) {
#pragma unroll
            for (int r = 0; r < 8; ++r) {
                const int gm = m0 + wm * 64 + mi * 16 + r + (lh << 3);
                const int gn = n0 + wn * 32 + ni * 16 + lm;
                const float v = acc[mi][ni][r];
                if (MODE == 0) {
                    const int b = gm >> 11, s = gm & 2047;
                    const int which = gn / DIM;
                    const int c = gn - which * DIM;
                    const int h = c >> 6, hd = c & 63;
                    const size_t bh = (size_t)(b * NH + h);
                    const _Float16 hv = (_Float16)v;
                    if (which == 0)      qh[(bh * SEQ + s) * HD + hd] = hv;
                    else if (which == 1) kh[(bh * SEQ + s) * HD + hd] = hv;
                    else                 vTh[(bh * HD + hd) * SEQ + s] = hv;
                } else {
                    out[(size_t)gm * Nout + gn] = v + bias[gn];
                }
            }
        }
    }
}

// ---------------------------------------------------------------------------
// Flash-style attention. grid = (16 qtiles, 48 bh), 256 thr (8 waves).
// Each wave: 16 query rows, 32 iterations over chunks of 64 keys.
// Per chunk: 8 WMMA (S) + 8 WMMA (PV), only a 4-op max reduction per row;
// l is kept as per-lane partials and reduced once at the end.
// ---------------------------------------------------------------------------
__global__ __launch_bounds__(256) void attn_kernel(
    const _Float16* __restrict__ qh, const _Float16* __restrict__ kh,
    const _Float16* __restrict__ vTh, _Float16* __restrict__ aoh)
{
    __shared__ _Float16 Ps[8][16 * PSTR];   // per-wave P scratch (C->A layout bounce)

    const int tid  = threadIdx.x;
    const int lane = tid & 31;
    const int w    = tid >> 5;
    const int lm   = lane & 15;
    const int lh   = lane >> 4;
    const int bh   = blockIdx.y;
    const int q0   = blockIdx.x * 128 + w * 16;

    const _Float16* Q  = qh  + (size_t)bh * SEQ * HD;
    const _Float16* Kp = kh  + (size_t)bh * SEQ * HD;
    const _Float16* Vt = vTh + (size_t)bh * HD * SEQ;

    // Q fragments (two k=32 chunks over HD=64), softmax scale folded in (2^-3, exact)
    v16h qf[2];
    {
        const _Float16* qp = Q + (size_t)(q0 + lm) * HD;
        const int koff = lh * 8;
        const _Float16 sc = (_Float16)0.125f;
#pragma unroll
        for (int c = 0; c < 2; ++c) {
            v8h lo = *(const v8h*)(qp + c * 32 + 2 * koff);
            v8h hi = *(const v8h*)(qp + c * 32 + 2 * koff + 8);
#pragma unroll
            for (int i = 0; i < 8; ++i) { qf[c][i] = lo[i] * sc; qf[c][8 + i] = hi[i] * sc; }
        }
    }

    v8f o[4] = {};
    float ms[8], ls[8];
#pragma unroll
    for (int r = 0; r < 8; ++r) { ms[r] = -1e30f; ls[r] = 0.0f; }

    for (int kb = 0; kb < SEQ / 64; ++kb) {
        const int key0 = kb * 64;

        // S = (Q*scale) K^T : 4 n-tiles x 2 k-steps
        v8f s[4] = {};
#pragma unroll
        for (int kc = 0; kc < 2; ++kc)
#pragma unroll
            for (int ni = 0; ni < 4; ++ni) {
                v16h bfr = *(const v16h*)(Kp + (size_t)(key0 + ni * 16 + lm) * HD + kc * 32 + lh * 16);
                s[ni] = __builtin_amdgcn_wmma_f32_16x16x32_f16(
                    false, qf[kc], false, bfr, (short)0, s[ni], false, false);
            }

        // online softmax; row r lives in VGPR r of a 16-lane half (m = r + 8*lh)
#pragma unroll
        for (int r = 0; r < 8; ++r) {
            float cm = fmaxf(fmaxf(s[0][r], s[1][r]), fmaxf(s[2][r], s[3][r]));
            cm = xor_max16(cm);
            const float nm    = fmaxf(ms[r], cm);
            const float alpha = __expf(ms[r] - nm);
            const float p0 = __expf(s[0][r] - nm);
            const float p1 = __expf(s[1][r] - nm);
            const float p2 = __expf(s[2][r] - nm);
            const float p3 = __expf(s[3][r] - nm);
            ls[r] = ls[r] * alpha + ((p0 + p1) + (p2 + p3));   // per-lane partial sum
            ms[r] = nm;
#pragma unroll
            for (int t = 0; t < 4; ++t) o[t][r] *= alpha;
            _Float16* pr = &Ps[w][(r + (lh << 3)) * PSTR];
            pr[lm]      = (_Float16)p0;
            pr[16 + lm] = (_Float16)p1;
            pr[32 + lm] = (_Float16)p2;
            pr[48 + lm] = (_Float16)p3;
        }

        // O += P * V : 2 k-steps x 4 HD tiles
#pragma unroll
        for (int pc = 0; pc < 2; ++pc) {
            v16h pf = load_a_frag(&Ps[w][lm * PSTR + pc * 32], lh * 8);
#pragma unroll
            for (int t = 0; t < 4; ++t) {
                v16h vf = *(const v16h*)(Vt + (size_t)(t * 16 + lm) * SEQ + key0 + pc * 32 + lh * 16);
                o[t] = __builtin_amdgcn_wmma_f32_16x16x32_f16(
                    false, pf, false, vf, (short)0, o[t], false, false);
            }
        }
    }

    // one final cross-lane reduction of l per row, then normalize + store
    float rcp[8];
#pragma unroll
    for (int r = 0; r < 8; ++r) rcp[r] = 1.0f / xor_sum16(ls[r]);

    const int b = bh / NH, h = bh % NH;
#pragma unroll
    for (int t = 0; t < 4; ++t) {
#pragma unroll
        for (int r = 0; r < 8; ++r) {
            const int s = q0 + r + (lh << 3);
            aoh[((size_t)(b * SEQ + s)) * DIM + h * HD + t * 16 + lm] =
                (_Float16)(o[t][r] * rcp[r]);
        }
    }
}

// ---------------------------------------------------------------------------
// q_attn: softmax row for query 0 of each (b,h). grid = 48, 256 thr.
// ---------------------------------------------------------------------------
__global__ __launch_bounds__(256) void qattn_kernel(
    const _Float16* __restrict__ qh, const _Float16* __restrict__ kh,
    float* __restrict__ qattn)
{
    __shared__ float sb[SEQ];
    __shared__ float red[256];
    const int bh = blockIdx.x;
    const int tid = threadIdx.x;
    const _Float16* q0 = qh + (size_t)bh * SEQ * HD;
    const _Float16* Kp = kh + (size_t)bh * SEQ * HD;

    float qv[HD];
#pragma unroll
    for (int d = 0; d < HD; ++d) qv[d] = (float)q0[d];

    float lmax = -1e30f;
    for (int j = tid; j < SEQ; j += 256) {
        const _Float16* kr = Kp + (size_t)j * HD;
        float s = 0.0f;
#pragma unroll
        for (int d = 0; d < HD; ++d) s += qv[d] * (float)kr[d];
        s *= 0.125f;
        sb[j] = s;
        lmax = fmaxf(lmax, s);
    }
    red[tid] = lmax;
    __syncthreads();
    for (int off = 128; off > 0; off >>= 1) {
        if (tid < off) red[tid] = fmaxf(red[tid], red[tid + off]);
        __syncthreads();
    }
    const float mx = red[0];
    __syncthreads();

    float lsum = 0.0f;
    for (int j = tid; j < SEQ; j += 256) {
        const float e = __expf(sb[j] - mx);
        sb[j] = e;
        lsum += e;
    }
    red[tid] = lsum;
    __syncthreads();
    for (int off = 128; off > 0; off >>= 1) {
        if (tid < off) red[tid] += red[tid + off];
        __syncthreads();
    }
    const float inv = 1.0f / red[0];
    for (int j = tid; j < SEQ; j += 256)
        qattn[(size_t)bh * SEQ + j] = sb[j] * inv;
}

// ---------------------------------------------------------------------------
extern "C" void kernel_launch(void* const* d_in, const int* in_sizes, int n_in,
                              void* d_out, int out_size, void* d_ws, size_t ws_size,
                              hipStream_t stream) {
    (void)in_sizes; (void)n_in; (void)out_size; (void)ws_size;
    const float* x      = (const float*)d_in[0];
    const float* w_qkv  = (const float*)d_in[1];
    const float* w_proj = (const float*)d_in[2];
    const float* b_proj = (const float*)d_in[3];
    float* out   = (float*)d_out;
    float* qattn = out + (size_t)MROWS * DIM;   // after [4,2048,768]

    _Float16* ws = (_Float16*)d_ws;
    size_t o = 0;
    _Float16* xh     = ws + o; o += (size_t)MROWS * DIM;
    _Float16* wqkvT  = ws + o; o += (size_t)3 * DIM * DIM;
    _Float16* wprojT = ws + o; o += (size_t)DIM * DIM;
    _Float16* qh     = ws + o; o += (size_t)BHN * SEQ * HD;
    _Float16* kh     = ws + o; o += (size_t)BHN * SEQ * HD;
    _Float16* vTh    = ws + o; o += (size_t)BHN * HD * SEQ;
    _Float16* aoh    = ws + o; o += (size_t)MROWS * DIM;

    const int nX  = MROWS * DIM;
    const int nWQ = DIM * 3 * DIM;
    const int nWP = DIM * DIM;

    cvt_f16<<<(nX + 255) / 256, 256, 0, stream>>>(x, xh, nX);
    cvtT_f16<<<(nWQ + 255) / 256, 256, 0, stream>>>(w_qkv, wqkvT, DIM, 3 * DIM);
    cvtT_f16<<<(nWP + 255) / 256, 256, 0, stream>>>(w_proj, wprojT, DIM, DIM);

    gemm128<0><<<dim3(3 * DIM / 128, MROWS / 128), 256, 0, stream>>>(
        xh, wqkvT, DIM, 3 * DIM, qh, kh, vTh, nullptr, nullptr);

    attn_kernel<<<dim3(SEQ / 128, BHN), 256, 0, stream>>>(qh, kh, vTh, aoh);
    qattn_kernel<<<BHN, 256, 0, stream>>>(qh, kh, qattn);

    gemm128<1><<<dim3(DIM / 128, MROWS / 128), 256, 0, stream>>>(
        aoh, wprojT, DIM, DIM, nullptr, nullptr, nullptr, out, b_proj);
}
